// ScaledDotProductAttention_14860586844446
// MI455X (gfx1250) — compile-verified, matrix operand order checked by reading, not run on hardware
//
#include <hip/hip_runtime.h>

// ---------------------------------------------------------------------------
// (B, L, DK) = (32, 1024, 64), P = 2L-1 = 2047
// d_out = [output (B*L*64 f32) | attn (B*L*L f32)]
// d_ws  = pre-swizzled bf16 fragments of k, pos, v (16 MB)
// ---------------------------------------------------------------------------
constexpr int Bb = 32;
constexpr int Ll = 1024;
constexpr int DK = 64;
constexpr int PP = 2 * Ll - 1;
constexpr float SCALE = 0.125f;      // 1/sqrt(64)
constexpr float NEGBIG = -1e30f;     // masked logit
constexpr int WAVES = 8;             // waves per 256-thread block

typedef __attribute__((ext_vector_type(16))) __bf16 v16bf;
typedef __attribute__((ext_vector_type(8)))  float  v8f;

// Workspace layout (bf16 elements):
//  ksw: [B][64 tiles][2 kblk][32 lanes][16]   (4 MB)
//  psw: [B][128 tiles][2 kblk][32 lanes][16]  (8 MB)
//  vsw: [B][32 ktiles][4 nt][32 lanes][16]    (4 MB)
constexpr size_t KSW_ELEMS = (size_t)Bb * 64  * 2 * 32 * 16;
constexpr size_t PSW_ELEMS = (size_t)Bb * 128 * 2 * 32 * 16;

// K index covered by element e of a lane's 16-element bf16 fragment.
__device__ __forceinline__ int kof(int lane, int e) {
  return ((lane & 16) ? 8 : 0) + e + ((e >= 8) ? 8 : 0);
}

__device__ __forceinline__ v8f vzero() {
  v8f z;
#pragma unroll
  for (int e = 0; e < 8; ++e) z[e] = 0.0f;
  return z;
}

__device__ __forceinline__ v8f wmma_bf16(v16bf a, v16bf b, v8f c) {
  return __builtin_amdgcn_wmma_f32_16x16x32_bf16(
      false, a, false, b, (short)0, c, false, false);
}

// ------------------------- swizzle kernels (prep) --------------------------
__global__ __launch_bounds__(256) void swz_rows_kernel(
    const float* __restrict__ src, __bf16* __restrict__ dst,
    int ntiles, int nrows) {                 // k: 64/1024, pos: 128/2047
  const int tid = threadIdx.x, wid = tid >> 5, lane = tid & 31;
  const int b = blockIdx.y;
  const int tile = blockIdx.x * WAVES + wid;
  int r = tile * 16 + (lane & 15);
  if (r > nrows - 1) r = nrows - 1;          // pos tail clamp (never gathered)
  const float* row = src + ((size_t)b * nrows + r) * DK;
  __bf16* out = dst + ((size_t)b * ntiles + tile) * (2 * 32 * 16);
#pragma unroll
  for (int kb = 0; kb < 2; ++kb) {
    v16bf f;
#pragma unroll
    for (int e = 0; e < 16; ++e) f[e] = (__bf16)row[kb * 32 + kof(lane, e)];
    *(v16bf*)(out + (size_t)(kb * 32 + lane) * 16) = f;
  }
}

__global__ __launch_bounds__(256) void swz_v_kernel(
    const float* __restrict__ src, __bf16* __restrict__ dst) {
  const int tid = threadIdx.x, wid = tid >> 5, lane = tid & 31;
  const int b = blockIdx.y;
  const int tile = blockIdx.x * WAVES + wid;         // 0..31 (32-row K tiles)
  const float* vb = src + ((size_t)b * Ll + tile * 32) * DK;
  __bf16* out = dst + ((size_t)b * 32 + tile) * (4 * 32 * 16);
#pragma unroll
  for (int nt = 0; nt < 4; ++nt) {
    const int colv = nt * 16 + (lane & 15);
    v16bf f;
#pragma unroll
    for (int e = 0; e < 16; ++e) f[e] = (__bf16)vb[(size_t)kof(lane, e) * DK + colv];
    *(v16bf*)(out + (size_t)(nt * 32 + lane) * 16) = f;
  }
}

// ------------------------------ main kernel --------------------------------
// U/V tile of the positional term: U[a,d] = q'_{i0+a} . pos[base+d]
__device__ __forceinline__ v8f pos_tile(const __bf16* psb, int base, int lane,
                                        v16bf ad0, v16bf ad1) {
  const __bf16* t = psb + (size_t)(base >> 4) * (2 * 32 * 16);
  v16bf b0 = *(const v16bf*)(t + (size_t)lane * 16);
  v16bf b1 = *(const v16bf*)(t + (size_t)(32 + lane) * 16);
  v8f c = vzero();
  c = wmma_bf16(ad0, b0, c);
  c = wmma_bf16(ad1, b1, c);
  return c;
}

// Raw QK^T 16x16 tile via two chained WMMAs.
__device__ __forceinline__ v8f qk_tile(const __bf16* ksb, int j0, int lane,
                                       v16bf aq0, v16bf aq1) {
  const __bf16* t = ksb + (size_t)(j0 >> 4) * (2 * 32 * 16);
  v16bf b0 = *(const v16bf*)(t + (size_t)lane * 16);
  v16bf b1 = *(const v16bf*)(t + (size_t)(32 + lane) * 16);
  v8f s = vzero();
  s = wmma_bf16(aq0, b0, s);
  s = wmma_bf16(aq1, b1, s);
  return s;
}

__global__ __launch_bounds__(256) void dpa_attn_kernel(
    const float* __restrict__ q, const unsigned char* __restrict__ mask,
    const __bf16* __restrict__ ksw, const __bf16* __restrict__ psw,
    const __bf16* __restrict__ vsw,
    float* __restrict__ out, float* __restrict__ attn) {
  __shared__ float pst[WAVES][16][33];   // per-wave P-tile transpose stage

  const int tid  = threadIdx.x;
  const int wid  = tid >> 5;
  const int lane = tid & 31;
  const int col  = lane & 15;
  const int half = lane >> 4;
  const int k0   = half ? 8 : 0;
  const int b    = blockIdx.y;
  const int i0   = (blockIdx.x * WAVES + wid) * 16;

  const float*  qb  = q   + (size_t)b * Ll * DK;
  const __bf16* ksb = ksw + (size_t)b * 64  * (2 * 32 * 16);
  const __bf16* psb = psw + (size_t)b * 128 * (2 * 32 * 16);
  const __bf16* vsb = vsw + (size_t)b * 32  * (4 * 32 * 16);
  const unsigned char* mrow = mask + (size_t)b * Ll * Ll + (size_t)i0 * Ll;

  // A fragments (built once): q rows i0..i0+15 and reversed dpa rows L-1-i.
  v16bf aq0, aq1, ad0, ad1;
  {
    const float* r0 = qb + (size_t)(i0 + col) * DK;
    const float* r1 = qb + (size_t)(Ll - 1 - i0 - col) * DK;
#pragma unroll
    for (int e = 0; e < 16; ++e) {
      int kq = kof(lane, e);
      aq0[e] = (__bf16)r0[kq];      aq1[e] = (__bf16)r0[32 + kq];
      ad0[e] = (__bf16)r1[kq];      ad1[e] = (__bf16)r1[32 + kq];
    }
  }

  // Column mask bits: bit t <=> columns [16t,16t+16) of this lane are masked.
  unsigned long long mbits = 0ull;
  for (int t = 0; t < 64; ++t)
    mbits |= (unsigned long long)(mrow[t * 16 + col] != 0) << t;

  // Loop-invariant gather controls for the diagonal DPA term.
  // Source-side pre-select: each source lane serves exactly one consumer per
  // register, so U-vs-V is decided before a single shuffle.
  int  sl[8];   // consumer-side: source lane for this element
  bool gs[8];   // source-side:   true -> serve U, false -> serve V
#pragma unroll
  for (int r = 0; r < 8; ++r) {
    const int a = r + 8 * half;
    sl[r] = (lane & 16) | ((a + col) & 15);
    gs[r] = ((col - a) & 15) < (16 - a);
  }

  // Gather + scale + mask applied to a raw QK tile.
  auto finish_tile = [&](v8f& s, const v8f& U, const v8f& V, bool msk) {
#pragma unroll
    for (int r = 0; r < 8; ++r) {
      float w  = gs[r] ? U[r] : V[r];      // pre-select at source lane
      float t2 = __shfl(w, sl[r], 32);     // one ds_bpermute per register
      float val = (s[r] + t2) * SCALE;
      s[r] = msk ? NEGBIG : val;
    }
  };

  // ---------------- pass 1: online row max / rescaled sum ----------------
  float mrun[8], srun[8];
#pragma unroll
  for (int r = 0; r < 8; ++r) { mrun[r] = -INFINITY; srun[r] = 0.0f; }

  v8f Vst = pos_tile(psb, i0, lane, ad0, ad1);     // U of tile j0 = 0
  for (int j0 = 0; j0 < Ll; j0 += 16) {
    v8f U = Vst;
    Vst = pos_tile(psb, i0 + j0 + 16, lane, ad0, ad1);
    v8f s = qk_tile(ksb, j0, lane, aq0, aq1);
    finish_tile(s, U, Vst, (mbits >> (j0 >> 4)) & 1);
#pragma unroll
    for (int r = 0; r < 8; ++r) {
      float e  = s[r];
      float nm = fmaxf(mrun[r], e);
      srun[r] = srun[r] * __expf(mrun[r] - nm) + __expf(e - nm);
      mrun[r] = nm;
    }
    if (j0 + 16 < Ll)
      __builtin_prefetch(ksb + (size_t)((j0 >> 4) + 1) * (2 * 32 * 16) +
                             (size_t)lane * 16, 0, 1);
  }
  // merge stats across the 16 lanes sharing each row
#pragma unroll
  for (int off = 1; off < 16; off <<= 1) {
#pragma unroll
    for (int r = 0; r < 8; ++r) {
      float om = __shfl_xor(mrun[r], off, 32);
      float os = __shfl_xor(srun[r], off, 32);
      float nm = fmaxf(mrun[r], om);
      srun[r] = srun[r] * __expf(mrun[r] - nm) + os * __expf(om - nm);
      mrun[r] = nm;
    }
  }
  float rinv[8];
#pragma unroll
  for (int r = 0; r < 8; ++r) rinv[r] = 1.0f / srun[r];

  // ------------- pass 2: recompute, write attn, accumulate O -------------
  v8f O0 = vzero(), O1 = vzero(), O2 = vzero(), O3 = vzero();
  float* ab = attn + (size_t)b * Ll * Ll;
  Vst = pos_tile(psb, i0, lane, ad0, ad1);
  for (int j0 = 0; j0 < Ll; j0 += 32) {            // pairs of 16-col tiles
#pragma unroll
    for (int t = 0; t < 2; ++t) {
      int jt = j0 + t * 16;
      v8f U = Vst;
      Vst = pos_tile(psb, i0 + jt + 16, lane, ad0, ad1);
      v8f s = qk_tile(ksb, jt, lane, aq0, aq1);
      finish_tile(s, U, Vst, (mbits >> (jt >> 4)) & 1);
#pragma unroll
      for (int r = 0; r < 8; ++r) {
        float p = __expf(s[r] - mrun[r]) * rinv[r];
        int row = r + 8 * half;
        ab[(size_t)(i0 + row) * Ll + jt + col] = p;   // attn output (once)
        pst[wid][row][t * 16 + col] = p;              // stage for transpose
      }
    }
    asm volatile("s_wait_dscnt 0" ::: "memory");      // stores visible in-wave

    // Rebuild P (16x32) as bf16 A-fragment: lane = row, K-striped reads.
    v16bf pa;
    {
      const float* rp = &pst[wid][col][0];
#pragma unroll
      for (int e = 0; e < 8; ++e) {
        pa[e]     = (__bf16)rp[k0 + e];
        pa[8 + e] = (__bf16)rp[k0 + 16 + e];
      }
    }
    asm volatile("s_wait_dscnt 0" ::: "memory");      // reads before next WAR

    // O(16x64) += P(16x32) x V(32x64): four pre-swizzled V fragments.
    const __bf16* vt = vsb + (size_t)(j0 >> 5) * (4 * 32 * 16);
#pragma unroll
    for (int nt = 0; nt < 4; ++nt) {
      v16bf bv = *(const v16bf*)(vt + (size_t)(nt * 32 + lane) * 16);
      v8f& O = (nt == 0) ? O0 : (nt == 1) ? O1 : (nt == 2) ? O2 : O3;
      O = wmma_bf16(pa, bv, O);
    }
  }

  // write O tiles (row = r + 8*half, col = nt*16 + col)
#pragma unroll
  for (int nt = 0; nt < 4; ++nt) {
    const v8f& O = (nt == 0) ? O0 : (nt == 1) ? O1 : (nt == 2) ? O2 : O3;
#pragma unroll
    for (int r = 0; r < 8; ++r) {
      out[(size_t)(b * Ll + i0 + r + 8 * half) * DK + nt * 16 + col] = O[r];
    }
  }
}

extern "C" void kernel_launch(void* const* d_in, const int* in_sizes, int n_in,
                              void* d_out, int out_size, void* d_ws, size_t ws_size,
                              hipStream_t stream) {
  const float* q   = (const float*)d_in[0];
  const float* k   = (const float*)d_in[1];
  const float* v   = (const float*)d_in[2];
  const float* pos = (const float*)d_in[3];
  const unsigned char* mask = (const unsigned char*)d_in[4];  // numpy bool, 1B

  float* out  = (float*)d_out;                   // (B, L, DK)
  float* attn = out + (size_t)Bb * Ll * DK;      // (B, L, L)

  __bf16* ksw = (__bf16*)d_ws;                   // 16 MB total scratch
  __bf16* psw = ksw + KSW_ELEMS;
  __bf16* vsw = psw + PSW_ELEMS;

  // Preprocess: convert + swizzle k, pos, v into bf16 WMMA fragments.
  hipLaunchKernelGGL(swz_rows_kernel, dim3(64 / WAVES, Bb), dim3(256), 0, stream,
                     k, ksw, 64, Ll);
  hipLaunchKernelGGL(swz_rows_kernel, dim3(128 / WAVES, Bb), dim3(256), 0, stream,
                     pos, psw, 128, PP);
  hipLaunchKernelGGL(swz_v_kernel, dim3(32 / WAVES, Bb), dim3(256), 0, stream,
                     v, vsw);

  hipLaunchKernelGGL(dpa_attn_kernel, dim3(Ll / 16 / WAVES, Bb), dim3(256), 0,
                     stream, q, mask, ksw, psw, vsw, out, attn);
}
